// NetGINConv_28226525070249
// MI455X (gfx1250) — matrix operand order repaired and checked by reading, not compiled, not run on hardware
//
#include <hip/hip_runtime.h>
#include <hip/hip_bf16.h>
#include <math.h>

// ---------------------------------------------------------------------------
// NetGINConv for MI455X (gfx1250, wave32).
// Dense contractions go through one generic v_wmma_f32_16x16x32_f16 GEMM.
// Operands are pre-packed into zero-padded f16 buffers so the WMMA K-loop is
// branch-free: 4x global_load_b128 + 1x v_wmma per 32-K chunk.
// Edge aggregation uses global f32 atomics; CBAM is wave-per-node.
// ---------------------------------------------------------------------------

#define N_NODES 50000
#define N_EDGES 800000

typedef __attribute__((ext_vector_type(16))) _Float16 v16h;
typedef __attribute__((ext_vector_type(8)))  float    v8f;
typedef __attribute__((ext_vector_type(4)))  int      v4i;
typedef __attribute__((ext_vector_type(8)))  int      v8i;

__device__ __forceinline__ float wsum(float v) {
#pragma unroll
  for (int o = 16; o > 0; o >>= 1) v += __shfl_xor(v, o, 32);
  return v;
}
__device__ __forceinline__ float wmaxr(float v) {
#pragma unroll
  for (int o = 16; o > 0; o >>= 1) v = fmaxf(v, __shfl_xor(v, o, 32));
  return v;
}
__device__ __forceinline__ float sigm(float x) { return 1.0f / (1.0f + __expf(-x)); }
__device__ __forceinline__ float lrelu(float x) { return x > 0.0f ? x : 0.01f * x; }

// ---------------------------------------------------------------------------
// Zero-padded f32 -> f16 pack: src [R,K] row-major -> dst [Rp,Kp] f16,
// rows >= R and cols >= K are zero. Rp mult of 64 (rows) / 16 (cols of W),
// Kp mult of 32, so the GEMM below never needs bounds checks on loads.
// ---------------------------------------------------------------------------
__global__ void pack_f16_k(const float* __restrict__ src, _Float16* __restrict__ dst,
                           int R, int K, int Rp, int Kp) {
  size_t t = (size_t)blockIdx.x * blockDim.x + threadIdx.x;
  size_t total = (size_t)Rp * Kp;
  if (t >= total) return;
  int k = (int)(t % Kp);
  int r = (int)(t / Kp);
  float v = (r < R && k < K) ? src[(size_t)r * K + k] : 0.0f;
  dst[t] = (_Float16)v;
}

// ---------------------------------------------------------------------------
// GEMM: C[M,Nout] = act(A[M,K] @ W[Nout,K]^T + bias), operands pre-packed f16.
// Block = 128 threads = 4 waves; each wave owns one 16x16 C tile.
// Fragment layouts (CDNA5 16-bit WMMA, wave32):
//   A 16x32: lane half h, VGPR j<4 -> k=2j+8h, j>=4 -> k=16+2(j-4)+8h
//            => two contiguous 16B runs per lane: [k0+8h, +8) and [k0+16+8h, +8)
//   B 32x16: lane half h, VGPR j<4 -> k=2j+8h, j>=4 -> k=16+2(j-4)+8h
//            => two contiguous 16B runs per lane: [k0+8h, +8) and [k0+16+8h, +8)
//   C/D:     lane n = L%16, VGPR r -> row r + 8*(L/16)
// EXEC is all-ones at every WMMA (only wave-uniform early exit before loop).
// ---------------------------------------------------------------------------
template <bool RELU>
__global__ __launch_bounds__(128) void gemm_wmma(
    const _Float16* __restrict__ A, const _Float16* __restrict__ W,
    const float* __restrict__ bias, float* __restrict__ C,
    int M, int Kp, int Nout) {
  const int lane = threadIdx.x & 31;
  const int wav  = threadIdx.x >> 5;
  const int tm   = (blockIdx.x * 4 + wav) * 16;  // wave-uniform
  const int tn   = blockIdx.y * 16;
  if (tm >= M) return;  // whole-wave uniform exit (pure-padding tile)

  const int half = lane >> 4;
  const int l16  = lane & 15;

  const _Float16* Arow = A + (size_t)(tm + l16) * Kp + 8 * half;   // in-bounds: padded
  const _Float16* Brow = W + (size_t)(tn + l16) * Kp + 8 * half;   // in-bounds: padded

  v8f acc = {};
  for (int k0 = 0; k0 < Kp; k0 += 32) {
    v4i a0 = *(const v4i*)(Arow + k0);
    v4i a1 = *(const v4i*)(Arow + k0 + 16);
    v4i b0 = *(const v4i*)(Brow + k0);
    v4i b1 = *(const v4i*)(Brow + k0 + 16);
    v16h av = __builtin_bit_cast(v16h, __builtin_shufflevector(a0, a1, 0, 1, 2, 3, 4, 5, 6, 7));
    v16h bv = __builtin_bit_cast(v16h, __builtin_shufflevector(b0, b1, 0, 1, 2, 3, 4, 5, 6, 7));
    acc = __builtin_amdgcn_wmma_f32_16x16x32_f16(false, av, false, bv,
                                                 (short)0, acc, false, false);
  }
  const int  ncol = tn + l16;
  const float bb  = (ncol < Nout) ? bias[ncol] : 0.0f;
#pragma unroll
  for (int r = 0; r < 8; ++r) {
    int m = tm + r + 8 * half;
    if (m < M && ncol < Nout) {
      float v = acc[r] + bb;
      if (RELU) v = lrelu(v);
      C[(size_t)m * Nout + ncol] = v;
    }
  }
}

// ---------------------------------------------------------------------------
// CBAM (channel gate + spatial gate), wave-per-node. Element (c,i) of node n
// lives at in[n*C*HW + c*ics + i*iis]; output strides ocs/ois independent.
// ---------------------------------------------------------------------------
template <int C, int MID, int HW>
__global__ __launch_bounds__(256) void cbam_k(
    const float* __restrict__ in, const float* __restrict__ m1w,
    const float* __restrict__ m1b, const float* __restrict__ m2w,
    const float* __restrict__ m2b, const float* __restrict__ sc,
    const float* __restrict__ bng, const float* __restrict__ bnb,
    float* __restrict__ out, int nNodes,
    int ics, int iis, int ocs, int ois, int doL) {
  const int node = blockIdx.x * 8 + (threadIdx.x >> 5);
  if (node >= nNodes) return;  // wave-uniform
  const int lane = threadIdx.x & 31;
  constexpr int CPL = (C + 31) / 32;

  const float* base = in + (size_t)node * C * HW;
  float v[CPL][HW];
  int   cidx[CPL];
  bool  ok[CPL];
#pragma unroll
  for (int j = 0; j < CPL; ++j) {
    int c = lane + 32 * j;
    cidx[j] = c;
    ok[j]   = (c < C);
#pragma unroll
    for (int i = 0; i < HW; ++i)
      v[j][i] = ok[j] ? base[c * ics + i * iis] : 0.0f;
  }

  // per-channel stats over spatial dim: mean, max, logsumexp
  float st[3][CPL];
#pragma unroll
  for (int j = 0; j < CPL; ++j) {
    float mx = -3.4e38f, sm = 0.0f;
#pragma unroll
    for (int i = 0; i < HW; ++i) { mx = fmaxf(mx, v[j][i]); sm += v[j][i]; }
    float es = 0.0f;
#pragma unroll
    for (int i = 0; i < HW; ++i) es += __expf(v[j][i] - mx);
    st[0][j] = sm / (float)HW;
    st[1][j] = mx;
    st[2][j] = mx + __logf(es);
  }

  // shared MLP (C -> MID -> C) on each of the 3 stats, summed
  float att[CPL];
#pragma unroll
  for (int j = 0; j < CPL; ++j) att[j] = 0.0f;
  for (int s = 0; s < 3; ++s) {
    float hid[MID];
    for (int h = 0; h < MID; ++h) {
      float p = 0.0f;
#pragma unroll
      for (int j = 0; j < CPL; ++j)
        if (ok[j]) p += m1w[h * C + cidx[j]] * st[s][j];
      p = wsum(p) + m1b[h];
      hid[h] = p > 0.0f ? p : 0.0f;  // relu
    }
#pragma unroll
    for (int j = 0; j < CPL; ++j)
      if (ok[j]) {
        float o = m2b[cidx[j]];
        for (int h = 0; h < MID; ++h) o += m2w[cidx[j] * MID + h] * hid[h];
        att[j] += o;
      }
  }

  float cg[CPL][HW];
#pragma unroll
  for (int j = 0; j < CPL; ++j) {
    float g = sigm(att[j]);
#pragma unroll
    for (int i = 0; i < HW; ++i) cg[j][i] = v[j][i] * g;
  }

  // spatial gate: 7x7 conv pad 3 on HWx1 -> only kw=3 taps are in-bounds
  float chm[HW], cha[HW];
#pragma unroll
  for (int i = 0; i < HW; ++i) {
    float pm = -3.4e38f, ps = 0.0f;
#pragma unroll
    for (int j = 0; j < CPL; ++j)
      if (ok[j]) { pm = fmaxf(pm, cg[j][i]); ps += cg[j][i]; }
    chm[i] = wmaxr(pm);
    cha[i] = wsum(ps) / (float)C;
  }
  const float bnscale = bng[0] * rsqrtf(1.0f + 1e-5f);
  float gate[HW];
#pragma unroll
  for (int i = 0; i < HW; ++i) {
    float acc = 0.0f;
    for (int kh = 0; kh < 7; ++kh) {
      int p = i + kh - 3;
      if (p >= 0 && p < HW)
        acc += sc[kh * 7 + 3] * chm[p] + sc[49 + kh * 7 + 3] * cha[p];
    }
    gate[i] = sigm(acc * bnscale + bnb[0]);
  }

  float* ob = out + (size_t)node * C * HW;
#pragma unroll
  for (int j = 0; j < CPL; ++j)
    if (ok[j])
#pragma unroll
      for (int i = 0; i < HW; ++i) {
        float val = cg[j][i] * gate[i];
        if (doL) val = lrelu(val);
        ob[cidx[j] * ocs + i * ois] = val;
      }
}

// ---------------------------------------------------------------------------
// Elementwise / data-movement kernels
// ---------------------------------------------------------------------------
__global__ void zero_k(float* __restrict__ p, size_t n) {
  size_t t = (size_t)blockIdx.x * blockDim.x + threadIdx.x;
  if (t < n) p[t] = 0.0f;
}

// GIN sum aggregation: aggr[dst] += x1[src]; one thread per (edge, 4 channels)
__global__ void scatter_k(const float* __restrict__ x1, const int* __restrict__ ei,
                          float* __restrict__ ag) {
  size_t t = (size_t)blockIdx.x * blockDim.x + threadIdx.x;
  const size_t total = (size_t)N_EDGES * 30;
  if (t >= total) return;
  size_t e = t / 30;
  int    q = (int)(t % 30);
  int s = ei[e];
  int d = ei[N_EDGES + e];
  const float* sp = x1 + (size_t)s * 120 + q * 4;
  float*       dp = ag + (size_t)d * 120 + q * 4;
  atomicAdd(dp + 0, sp[0]);
  atomicAdd(dp + 1, sp[1]);
  atomicAdd(dp + 2, sp[2]);
  atomicAdd(dp + 3, sp[3]);
}

__global__ void hcomb_k(const float* __restrict__ x1, float* __restrict__ ag,
                        const float* __restrict__ eps) {
  size_t t = (size_t)blockIdx.x * blockDim.x + threadIdx.x;
  if (t >= (size_t)N_NODES * 120) return;
  ag[t] = (1.0f + eps[0]) * x1[t] + ag[t];
}

// faithful reshape(60,-1).T : y2[a*60+b] = z1_flat[b*N + a]
__global__ void permute_k(const float* __restrict__ z, float* __restrict__ y) {
  size_t t = (size_t)blockIdx.x * blockDim.x + threadIdx.x;
  if (t >= (size_t)N_NODES * 60) return;
  size_t a = t / 60;
  int    b = (int)(t % 60);
  y[t] = z[(size_t)b * N_NODES + a];
}

__global__ void addlrelu_k(const float* __restrict__ a, const float* __restrict__ b,
                           float* __restrict__ o, size_t n) {
  size_t t = (size_t)blockIdx.x * blockDim.x + threadIdx.x;
  if (t < n) o[t] = lrelu(a[t] + b[t]);
}

// im2col for conv_a: rows (n,i) i<5, cols (ci*4+kh*2+kw), inputs x | gc
__global__ void im2col_a_k(const float* __restrict__ x, const float* __restrict__ gc,
                           float* __restrict__ col) {
  size_t t = (size_t)blockIdx.x * blockDim.x + threadIdx.x;
  const size_t total = (size_t)N_NODES * 5 * 8;
  if (t >= total) return;
  int    cid = (int)(t & 7);
  size_t r   = t >> 3;
  int    i   = (int)(r % 5);
  size_t n   = r / 5;
  int ci = cid >> 2, kh = (cid >> 1) & 1, kw = cid & 1;
  size_t src = n * 30 + (size_t)(3 * i + kh) * 2 + kw;
  col[t] = (ci == 0) ? x[src] : gc[src];
}

// im2col for conv_b: ca is [N,5,64] position-major; rows (n,h) h<2, cols (c*2+kh)
__global__ void im2col_b_k(const float* __restrict__ ca, float* __restrict__ col) {
  size_t t = (size_t)blockIdx.x * blockDim.x + threadIdx.x;
  const size_t total = (size_t)N_NODES * 2 * 128;
  if (t >= total) return;
  int    cc = (int)(t & 127);
  size_t r  = t >> 7;
  int c = cc >> 1, kh = cc & 1;
  int h = (int)(r & 1);
  size_t n = r >> 1;
  col[t] = ca[n * 320 + (size_t)(2 * h + kh) * 64 + c];
}

// ---------------------------------------------------------------------------
static inline dim3 ewg(size_t n) { return dim3((unsigned)((n + 255) / 256)); }

// pack A and W to padded f16, then run the branch-free WMMA GEMM
static inline void gemm_full(bool relu, const float* A, const float* W,
                             const float* bias, float* C, int M, int K, int Nout,
                             _Float16* AH, _Float16* WH, hipStream_t s) {
  const int Kp = (K + 31) & ~31;
  const int Mp = (M + 63) & ~63;
  const int Np = (Nout + 15) & ~15;
  pack_f16_k<<<ewg((size_t)Mp * Kp), dim3(256), 0, s>>>(A, AH, M, K, Mp, Kp);
  pack_f16_k<<<ewg((size_t)Np * Kp), dim3(256), 0, s>>>(W, WH, Nout, K, Np, Kp);
  dim3 g(Mp / 64, Np / 16), blk(128);
  if (relu) gemm_wmma<true><<<g, blk, 0, s>>>(AH, WH, bias, C, M, Kp, Nout);
  else      gemm_wmma<false><<<g, blk, 0, s>>>(AH, WH, bias, C, M, Kp, Nout);
}

extern "C" void kernel_launch(void* const* d_in, const int* in_sizes, int n_in,
                              void* d_out, int out_size, void* d_ws, size_t ws_size,
                              hipStream_t stream) {
  (void)in_sizes; (void)n_in; (void)out_size;
  // -- inputs in setup_inputs() dict (insertion) order, recursively flattened --
  const float* x      = (const float*)d_in[0];   // [N,1,15,2]
  const float* x_real = (const float*)d_in[1];   // [N,60]
  const int*   ei     = (const int*)  d_in[2];   // [2,E]
  const float* fc_w   = (const float*)d_in[3];   // [120,60]
  const float* fc_b   = (const float*)d_in[4];
  const float* eps    = (const float*)d_in[5];
  const float* l1_w   = (const float*)d_in[6];   // [60,120]
  const float* l1_b   = (const float*)d_in[7];
  const float* l2_w   = (const float*)d_in[8];   // [30,60]
  const float* l2_b   = (const float*)d_in[9];
  const float* lc1_w  = (const float*)d_in[10];  // [60,120]
  const float* lc1_b  = (const float*)d_in[11];
  const float* c1m1w  = (const float*)d_in[12];  // lc1 cbam mlp1 [3,60]
  const float* c1m1b  = (const float*)d_in[13];
  const float* c1m2w  = (const float*)d_in[14];  // [60,3]
  const float* c1m2b  = (const float*)d_in[15];
  const float* c1sc   = (const float*)d_in[16];
  const float* c1bg   = (const float*)d_in[17];
  const float* c1bb   = (const float*)d_in[18];
  const float* lc2_w  = (const float*)d_in[19];  // [30,60]
  const float* lc2_b  = (const float*)d_in[20];
  const float* c2m1w  = (const float*)d_in[21];  // [1,30]
  const float* c2m1b  = (const float*)d_in[22];
  const float* c2m2w  = (const float*)d_in[23];  // [30,1]
  const float* c2m2b  = (const float*)d_in[24];
  const float* c2sc   = (const float*)d_in[25];
  const float* c2bg   = (const float*)d_in[26];
  const float* c2bb   = (const float*)d_in[27];
  const float* cva_w  = (const float*)d_in[28];  // [64,2,2,2] -> [64,8]
  const float* cva_b  = (const float*)d_in[29];
  const float* cam1w  = (const float*)d_in[30];  // cbam_a mlp1 [4,64]
  const float* cam1b  = (const float*)d_in[31];
  const float* cam2w  = (const float*)d_in[32];  // [64,4]
  const float* cam2b  = (const float*)d_in[33];
  const float* casc   = (const float*)d_in[34];
  const float* cabg   = (const float*)d_in[35];
  const float* cabb   = (const float*)d_in[36];
  const float* cvb_w  = (const float*)d_in[37];  // [128,64,2,1] -> [128,128]
  const float* cvb_b  = (const float*)d_in[38];
  const float* cbm1w  = (const float*)d_in[39];  // [8,128]
  const float* cbm1b  = (const float*)d_in[40];
  const float* cbm2w  = (const float*)d_in[41];  // [128,8]
  const float* cbm2b  = (const float*)d_in[42];
  const float* cbsc   = (const float*)d_in[43];
  const float* cbbg   = (const float*)d_in[44];
  const float* cbbb   = (const float*)d_in[45];
  const float* cvc_w  = (const float*)d_in[46];  // [256,128,2] -> [256,256]
  const float* cvc_b  = (const float*)d_in[47];
  const float* ccm1w  = (const float*)d_in[48];  // [16,256]
  const float* ccm1b  = (const float*)d_in[49];
  const float* ccm2w  = (const float*)d_in[50];  // [256,16]
  const float* ccm2b  = (const float*)d_in[51];
  const float* ccsc   = (const float*)d_in[52];
  const float* ccbg   = (const float*)d_in[53];
  const float* ccbb   = (const float*)d_in[54];
  const float* cvd_w  = (const float*)d_in[55];  // [50,256]
  const float* cvd_b  = (const float*)d_in[56];
  float* out = (float*)d_out;

  // -- workspace layout (floats): two 16M-float ping-pong regions + persistents
  //    + f16 staging areas for the GEMM operands.
  const size_t TOT = 42100000;  // ~168 MB
  if (ws_size < TOT * sizeof(float)) return;
  float*    ws   = (float*)d_ws;
  float*    RA   = ws;             // 16,000,000 floats
  float*    RB   = ws + 16000000;  // 16,000,000 floats
  float*    GC   = ws + 32000000;  //  1,500,000 (combined GIN out)
  float*    COLA = ws + 33500000;  //  2,000,000
  _Float16* AH   = (_Float16*)(ws + 35500000);  // 13,000,000 halves max
  _Float16* WH   = (_Float16*)(ws + 42000000);  //     65,536 halves max
  // front-phase buffers overlaid on RA / RB (dead before the CNN tail starts)
  float* X1 = RA;             // [N,120]
  float* HB = RA + 6000000;   // aggr, then h in place
  float* T1 = RA + 12000000;  // [N,60]
  float* G1 = RB;             // [N,30]
  float* Y1 = RB + 1500000;   // [N,60]
  float* Z1 = RB + 4500000;   // [N,60]
  float* Y2 = RB + 7500000;   // [N,60]
  float* Q  = RB + 10500000;  // [N,30]
  float* G3 = RB + 12000000;  // [N,30]

  const int N = N_NODES;
  dim3 blk(256);

  // 1. x1 = lrelu(fc(x_real))                                [N,120]
  gemm_full(true, x_real, fc_w, fc_b, X1, N, 60, 120, AH, WH, stream);

  // 2. aggr = segment_sum(x1[src] -> dst)   (re-zeroed every call for replay)
  zero_k<<<ewg((size_t)N * 120), blk, 0, stream>>>(HB, (size_t)N * 120);
  scatter_k<<<ewg((size_t)N_EDGES * 30), blk, 0, stream>>>(X1, ei, HB);

  // 3. h = (1+eps)*x1 + aggr  (in place in HB)
  hcomb_k<<<ewg((size_t)N * 120), blk, 0, stream>>>(X1, HB, eps);

  // 4-5. ginn: lrelu(l2(lrelu(l1(h))))                       [N,30]
  gemm_full(true, HB, l1_w, l1_b, T1, N, 120, 60, AH, WH, stream);
  gemm_full(true, T1, l2_w, l2_b, G1, N, 60, 30, AH, WH, stream);

  // 6-10. ginn3: lc1 -> cbam+lrelu -> permute -> lc2 -> cbam+lrelu
  gemm_full(false, HB, lc1_w, lc1_b, Y1, N, 120, 60, AH, WH, stream);
  cbam_k<60, 3, 1><<<dim3((N + 7) / 8), blk, 0, stream>>>(
      Y1, c1m1w, c1m1b, c1m2w, c1m2b, c1sc, c1bg, c1bb, Z1, N, 1, 1, 1, 1, 1);
  permute_k<<<ewg((size_t)N * 60), blk, 0, stream>>>(Z1, Y2);
  gemm_full(false, Y2, lc2_w, lc2_b, Q, N, 60, 30, AH, WH, stream);
  cbam_k<30, 1, 1><<<dim3((N + 7) / 8), blk, 0, stream>>>(
      Q, c2m1w, c2m1b, c2m2w, c2m2b, c2sc, c2bg, c2bb, G3, N, 1, 1, 1, 1, 1);

  // 11. x1' = lrelu(ginn + ginn3)                            [N,30] -> GC
  addlrelu_k<<<ewg((size_t)N * 30), blk, 0, stream>>>(G1, G3, GC, (size_t)N * 30);

  // 12-13. conv_a as GEMM over [N*5, 8] (RA front buffers now dead)
  im2col_a_k<<<ewg((size_t)N * 40), blk, 0, stream>>>(x, GC, COLA);
  gemm_full(true, COLA, cva_w, cva_b, RA, N * 5, 8, 64, AH, WH, stream);  // RA=[N,5,64]

  // 14. cbam_a (C=64, HW=5, position-major in/out)
  cbam_k<64, 4, 5><<<dim3((N + 7) / 8), blk, 0, stream>>>(
      RA, cam1w, cam1b, cam2w, cam2b, casc, cabg, cabb, RB, N, 1, 64, 1, 64, 0);

  // 15-16. conv_b as GEMM over [N*2, 128]
  im2col_b_k<<<ewg((size_t)N * 256), blk, 0, stream>>>(RB, RA);              // RA=[N*2,128]
  gemm_full(true, RA, cvb_w, cvb_b, RB, N * 2, 128, 128, AH, WH, stream);    // RB=[N,2,128]

  // 17. cbam_b (C=128, HW=2); output channel-major so conv_c reads contiguous (c,kh)
  cbam_k<128, 8, 2><<<dim3((N + 7) / 8), blk, 0, stream>>>(
      RB, cbm1w, cbm1b, cbm2w, cbm2b, cbsc, cbbg, cbbb, RA, N, 1, 128, 2, 1, 0);

  // 18. conv_c: pure GEMM [N,256]x[256,256]
  gemm_full(true, RA, cvc_w, cvc_b, RB, N, 256, 256, AH, WH, stream);        // RB=[N,256]

  // 19. cbam_c (C=256, HW=1)
  cbam_k<256, 16, 1><<<dim3((N + 7) / 8), blk, 0, stream>>>(
      RB, ccm1w, ccm1b, ccm2w, ccm2b, ccsc, ccbg, ccbb, RA, N, 1, 1, 1, 1, 0);

  // 20. conv_d + final lrelu -> d_out [N,50]
  gemm_full(true, RA, cvd_w, cvd_b, out, N, 256, 50, AH, WH, stream);
}
